// ContextMem_75840532513232
// MI455X (gfx1250) — matrix operand order repaired and checked by reading, not compiled, not run on hardware
//
#include <hip/hip_runtime.h>

// Problem constants (fixed by the reference: B=2, T=2048, C=1024, nh=16, hs=64)
#define B_  2
#define T_  2048
#define C_  1024
#define NH_ 16
#define HS_ 64
#define M_  (B_ * T_)

typedef __attribute__((ext_vector_type(16))) __bf16 v16bf;
typedef __attribute__((ext_vector_type(8)))  float  v8f;

union FragU { unsigned int d[8]; v16bf v; };

__device__ __forceinline__ unsigned short f2bf(float f) {
  unsigned int u = __float_as_uint(f);
  u += 0x7FFFu + ((u >> 16) & 1u);            // round-to-nearest-even
  return (unsigned short)(u >> 16);
}

// gfx1250 async copy: cache/memory -> LDS without touching VGPRs (ASYNCcnt).
__device__ __forceinline__ void async_copy_b128(unsigned lds_addr, const void* gaddr) {
  asm volatile("global_load_async_to_lds_b128 %0, %1, off"
               :: "v"(lds_addr), "v"(gaddr) : "memory");
}
__device__ __forceinline__ void wait_async_le4() {
  asm volatile("s_wait_asynccnt 0x4" ::: "memory");
}
__device__ __forceinline__ void wait_async_le0() {
  asm volatile("s_wait_asynccnt 0x0" ::: "memory");
}

// ---------------------------------------------------------------------------
// f32 -> bf16 elementwise convert
// ---------------------------------------------------------------------------
__global__ void cm_f32_to_bf16(const float* __restrict__ s,
                               unsigned short* __restrict__ d, int n) {
  int i = blockIdx.x * 256 + threadIdx.x;
  if (i < n) d[i] = f2bf(s[i]);
}

// ---------------------------------------------------------------------------
// Tiled bf16 WMMA GEMM:  Out[m,n] = sum_k A[m,k] * W[n,k]   (i.e. A @ W^T)
// Workgroup tile 128(M) x 128(N), K-chunks of 32, 8 waves; each wave owns a
// 32x64 slab -> 8 WMMAs per wave per chunk. Tiles double-buffered in LDS via
// global_load_async_to_lds_b128: next chunk's copies overlap this chunk's
// WMMAs, synchronized with s_wait_asynccnt 4 (newest 4 stay in flight).
// MODE 0: row-major MxN f32.  MODE 1: scatter to (B,NH,T,HS) f32.
// ---------------------------------------------------------------------------
#define GT_MN 128
#define GT_K  32
#define LDS_STRIDE 40   // padded row stride (elems) to dodge bank conflicts

template <int MODE>
__global__ __launch_bounds__(256)
void cm_gemm_bf16(const unsigned short* __restrict__ A,
                  const unsigned short* __restrict__ W,
                  float* __restrict__ Out, int Ktot, int Ntot)
{
  __shared__ __align__(16) unsigned short As[2][GT_MN * LDS_STRIDE];
  __shared__ __align__(16) unsigned short Bs[2][GT_MN * LDS_STRIDE];

  const int tid  = threadIdx.x;
  const int lane = tid & 31;
  const int wave = tid >> 5;        // 0..7
  const int wm   = wave & 3;        // 32-row slab
  const int wn   = wave >> 2;       // 64-col slab
  const int half = lane >> 4;       // upper/lower 16-lane group
  const int l16  = lane & 15;
  const int m0   = blockIdx.y * GT_MN;
  const int n0   = blockIdx.x * GT_MN;

  // staging assignment: each thread moves 32B of A tile + 32B of B tile
  const int srow = tid >> 1, sseg = (tid & 1) * 16;
  const unsigned a_lds0 = (unsigned)(size_t)&As[0][srow * LDS_STRIDE + sseg];
  const unsigned a_lds1 = (unsigned)(size_t)&As[1][srow * LDS_STRIDE + sseg];
  const unsigned b_lds0 = (unsigned)(size_t)&Bs[0][srow * LDS_STRIDE + sseg];
  const unsigned b_lds1 = (unsigned)(size_t)&Bs[1][srow * LDS_STRIDE + sseg];
  const unsigned short* agp = A + (size_t)(m0 + srow) * Ktot + sseg;
  const unsigned short* bgp = W + (size_t)(n0 + srow) * Ktot + sseg;

  auto issue = [&](int buf, int k0) {
    const unsigned al = buf ? a_lds1 : a_lds0;
    const unsigned bl = buf ? b_lds1 : b_lds0;
    async_copy_b128(al,      agp + k0);
    async_copy_b128(al + 16, agp + k0 + 8);
    async_copy_b128(bl,      bgp + k0);
    async_copy_b128(bl + 16, bgp + k0 + 8);
  };

  v8f acc[2][4];
#pragma unroll
  for (int a = 0; a < 2; ++a)
#pragma unroll
    for (int b = 0; b < 4; ++b)
#pragma unroll
      for (int i = 0; i < 8; ++i) acc[a][b][i] = 0.0f;

  issue(0, 0);
  int cur = 0;
  for (int k0 = 0; k0 < Ktot; k0 += GT_K) {
    const bool more = (k0 + GT_K) < Ktot;
    if (more) {            // overlap next chunk's copies with this chunk's WMMAs
      issue(cur ^ 1, k0 + GT_K);
      wait_async_le4();    // this chunk's 4 copies (older) are complete
    } else {
      wait_async_le0();
    }
    __syncthreads();

    const unsigned short* Asb = As[cur];
    const unsigned short* Bsb = Bs[cur];

    // A fragments, contiguous K order matching B (lanes 0-15: K 0-15; lanes 16-31: K 16-31)
    v16bf af[2], bfv[4];
#pragma unroll
    for (int mt = 0; mt < 2; ++mt) {
      FragU u;
      const int mrow = wm * 32 + mt * 16 + l16;
#pragma unroll
      for (int j = 0; j < 8; ++j) {
        const int kk = half * 16 + j * 2;
        u.d[j] = *(const unsigned int*)&Asb[mrow * LDS_STRIDE + kk];
      }
      af[mt] = u.v;
    }
    // B fragments, contiguous K order matching A (lanes 0-15: K 0-15; lanes 16-31: K 16-31)
#pragma unroll
    for (int nt = 0; nt < 4; ++nt) {
      FragU u;
      const int nrow = wn * 64 + nt * 16 + l16;
#pragma unroll
      for (int j = 0; j < 8; ++j)
        u.d[j] = *(const unsigned int*)&Bsb[nrow * LDS_STRIDE + half * 16 + 2 * j];
      bfv[nt] = u.v;
    }
#pragma unroll
    for (int mt = 0; mt < 2; ++mt)
#pragma unroll
      for (int nt = 0; nt < 4; ++nt)
        acc[mt][nt] = __builtin_amdgcn_wmma_f32_16x16x32_bf16(
            false, af[mt], false, bfv[nt], (short)0, acc[mt][nt], false, false);
    __syncthreads();
    cur ^= 1;
  }

  // epilogue: C/D layout -> (row = i + 8*half, col = l16); straight-line per MODE
#pragma unroll
  for (int mt = 0; mt < 2; ++mt)
#pragma unroll
    for (int nt = 0; nt < 4; ++nt) {
      const int n = n0 + wn * 64 + nt * 16 + l16;
      const int mbase = m0 + wm * 32 + mt * 16 + 8 * half;
      if (MODE == 0) {
        float* op = Out + (size_t)mbase * Ntot + n;
#pragma unroll
        for (int i = 0; i < 8; ++i) op[(size_t)i * Ntot] = acc[mt][nt][i];
      } else {  // scatter to (B, NH, T, HS): m = b*T + t, n = h*HS + d
        const int h = n >> 6, d = n & (HS_ - 1);
#pragma unroll
        for (int i = 0; i < 8; ++i) {
          const int m = mbase + i;
          const int b = m >> 11, t = m & (T_ - 1);
          Out[(((size_t)(b * NH_ + h)) * T_ + t) * HS_ + d] = acc[mt][nt][i];
        }
      }
    }
}

// ---------------------------------------------------------------------------
// Causal exponential-decay mix as an O(T) recurrence:
//   k[t] += exp(-|beta_h|*10) * k[t-1]     (2048 independent (b,h,d) lanes)
// ---------------------------------------------------------------------------
__global__ void cm_decay(float* __restrict__ K, const float* __restrict__ beta) {
  const int idx = blockIdx.x * 256 + threadIdx.x;       // 0..2047
  const int d = idx & (HS_ - 1);
  const int h = (idx >> 6) & (NH_ - 1);
  const int b = idx >> 10;
  const float alpha = __expf(-fabsf(beta[h]) * 10.0f);
  float* p = K + ((size_t)(b * NH_ + h) * T_) * HS_ + d;
  float prev = p[0];
  for (int t = 1; t < T_; ++t) {
    const float cur = p[(size_t)t * HS_] + alpha * prev;
    p[(size_t)t * HS_] = cur;
    prev = cur;
  }
}

// ---------------------------------------------------------------------------
// Row L2-normalize + per-head scale; one wave per (b,h,t) row of 64.
// ---------------------------------------------------------------------------
__global__ __launch_bounds__(256)
void cm_knorm(const float* __restrict__ K, const float* __restrict__ key_scale,
              unsigned short* __restrict__ kn) {
  const int row  = blockIdx.x * 8 + (threadIdx.x >> 5);
  const int lane = threadIdx.x & 31;
  const int h    = (row >> 11) & (NH_ - 1);
  const float* src = K + (size_t)row * HS_;
  float a = src[lane], b = src[lane + 32];
  float ss = a * a + b * b;
#pragma unroll
  for (int m = 16; m >= 1; m >>= 1) ss += __shfl_xor(ss, m, 32);
  const float KSM = 11.090339660644531f;   // log(2^16 - 1)
  const float sc = __expf(fminf(10.0f * key_scale[h], KSM)) / (sqrtf(ss) + 1e-10f);
  kn[(size_t)row * HS_ + lane]      = f2bf(a * sc);
  kn[(size_t)row * HS_ + lane + 32] = f2bf(b * sc);
}

__global__ __launch_bounds__(256)
void cm_vnorm(const float* __restrict__ V, const float* __restrict__ v_coef,
              const float* __restrict__ val_scale, unsigned short* __restrict__ vn) {
  const int row  = blockIdx.x * 8 + (threadIdx.x >> 5);
  const int lane = threadIdx.x & 31;
  const int h    = (row >> 11) & (NH_ - 1);
  const int t    = row & (T_ - 1);
  const float* src = V + (size_t)row * HS_;
  const float vc = v_coef[h];
  float a = src[lane], b = src[lane + 32];
  float an = 0.0f, bn = 0.0f;
  if (t < T_ - 1) { an = src[HS_ + lane]; bn = src[HS_ + lane + 32]; }
  a = (1.0f - vc) * an + vc * a;          // v = (1-c)*v[t+1] + c*v[t]
  b = (1.0f - vc) * bn + vc * b;
  float ss = a * a + b * b;
#pragma unroll
  for (int m = 16; m >= 1; m >>= 1) ss += __shfl_xor(ss, m, 32);
  const float sc = __expf(10.0f * val_scale[h]) / (sqrtf(ss) + 1e-10f);
  vn[(size_t)row * HS_ + lane]      = f2bf(a * sc);
  vn[(size_t)row * HS_ + lane + 32] = f2bf(b * sc);
}

// ---------------------------------------------------------------------------
// Flash attention, one wave per (b, h, 16-query block).
// Output row t (t>=1) attends keys 0..t-1 with q = kn[t]; row 0 = zeros.
// All S-phase fragments are gathered before the WMMAs so loads clause up;
// causal masking is skipped (wave-uniform branch) for fully-valid key blocks.
// ---------------------------------------------------------------------------
__global__ __launch_bounds__(32)
void cm_attn(const unsigned short* __restrict__ kn,
             const unsigned short* __restrict__ vn,
             unsigned short* __restrict__ ybf) {
  const int bh   = blockIdx.x >> 7;              // 128 q-blocks per (b,h)
  const int t0   = (blockIdx.x & 127) * 16;
  const int b    = bh >> 4, h = bh & (NH_ - 1);
  const int lane = threadIdx.x;
  const int half = lane >> 4, l16 = lane & 15;
  const unsigned short* Kbh = kn + (size_t)bh * T_ * HS_;
  const unsigned short* Vbh = vn + (size_t)bh * T_ * HS_;

  __shared__ __align__(16) unsigned short Ps[16 * LDS_STRIDE];
  __shared__ __align__(16) unsigned short VT[64 * LDS_STRIDE];   // V^T [d][k]

  // Q fragments (16 rows x 64 hs -> two K=32 A-fragments), gathered from global
  v16bf qf[2];
#pragma unroll
  for (int kk = 0; kk < 2; ++kk) {
    FragU u;
    const unsigned short* qrow = Kbh + (size_t)(t0 + l16) * HS_ + kk * 32;
#pragma unroll
    for (int j = 0; j < 8; ++j) {
      const int K = half * 16 + j * 2;
      u.d[j] = *(const unsigned int*)(qrow + K);
    }
    qf[kk] = u.v;
  }

  v8f o[4];
  float rm[8], rs[8];
#pragma unroll
  for (int dt = 0; dt < 4; ++dt)
#pragma unroll
    for (int i = 0; i < 8; ++i) o[dt][i] = 0.0f;
#pragma unroll
  for (int i = 0; i < 8; ++i) { rm[i] = -1e30f; rs[i] = 0.0f; }

  for (int j0 = 0; j0 <= t0 + 14; j0 += 32) {     // max key = t0+14
    // prefetch next key/value blocks (global_prefetch_b8)
    if (j0 + 32 <= t0 + 14) {
      __builtin_prefetch(Kbh + (size_t)(j0 + 32) * HS_ + 2 * lane, 0, 0);
      __builtin_prefetch(Vbh + (size_t)(j0 + 32) * HS_ + 2 * lane, 0, 0);
    }
    // stage V^T block: VT[d][k] = V[j0+k][d]
#pragma unroll 4
    for (int r = 0; r < 32; ++r) {
      const unsigned int pair =
          *(const unsigned int*)(Vbh + (size_t)(j0 + r) * HS_ + 2 * lane);
      VT[(2 * lane)     * LDS_STRIDE + r] = (unsigned short)(pair & 0xffffu);
      VT[(2 * lane + 1) * LDS_STRIDE + r] = (unsigned short)(pair >> 16);
    }

    // gather ALL key B-fragments first so the 8 b128 loads clause up
    FragU ub[2][2];
#pragma unroll
    for (int ct = 0; ct < 2; ++ct)
#pragma unroll
      for (int kk = 0; kk < 2; ++kk) {
        const unsigned short* krow =
            Kbh + (size_t)(j0 + ct * 16 + l16) * HS_ + kk * 32 + half * 16;
#pragma unroll
        for (int j = 0; j < 8; ++j)
          ub[ct][kk].d[j] = *(const unsigned int*)(krow + 2 * j);
      }

    // S = Q K^T : two 16x16 tiles, contraction over hs=64 (2 x K32 WMMA each)
    v8f s[2];
#pragma unroll
    for (int ct = 0; ct < 2; ++ct) {
#pragma unroll
      for (int i = 0; i < 8; ++i) s[ct][i] = 0.0f;
#pragma unroll
      for (int kk = 0; kk < 2; ++kk)
        s[ct] = __builtin_amdgcn_wmma_f32_16x16x32_bf16(
            false, qf[kk], false, ub[ct][kk].v, (short)0, s[ct], false, false);
    }

    // causal mask only on diagonal-straddling blocks (wave-uniform branch)
    const bool fullBlock = (j0 + 31) < t0;        // every key < every query row
    float bmax[8], bsum[8], rsc[8];
    if (fullBlock) {
#pragma unroll
      for (int i = 0; i < 8; ++i) bmax[i] = fmaxf(s[0][i], s[1][i]);
    } else {
#pragma unroll
      for (int i = 0; i < 8; ++i) {
        const int trow = t0 + i + 8 * half;
        float mx = -1e30f;
#pragma unroll
        for (int ct = 0; ct < 2; ++ct) {
          const int j = j0 + ct * 16 + l16;
          float v = s[ct][i];
          v = (j >= trow) ? -1e30f : v;           // valid keys: j <= trow-1
          s[ct][i] = v;
          mx = fmaxf(mx, v);
        }
        bmax[i] = mx;
      }
    }
#pragma unroll
    for (int i = 0; i < 8; ++i)
#pragma unroll
      for (int msk = 8; msk >= 1; msk >>= 1)
        bmax[i] = fmaxf(bmax[i], __shfl_xor(bmax[i], msk, 32));
#pragma unroll
    for (int i = 0; i < 8; ++i) {
      const float nm = fmaxf(rm[i], bmax[i]);
      rsc[i] = __expf(rm[i] - nm);
      rm[i] = nm;
      float ps = 0.0f;
#pragma unroll
      for (int ct = 0; ct < 2; ++ct) {
        const float p = __expf(s[ct][i] - nm);
        s[ct][i] = p;
        ps += p;
      }
      bsum[i] = ps;
    }
#pragma unroll
    for (int i = 0; i < 8; ++i) {
#pragma unroll
      for (int msk = 8; msk >= 1; msk >>= 1)
        bsum[i] += __shfl_xor(bsum[i], msk, 32);
      rs[i] = rs[i] * rsc[i] + bsum[i];
    }

    // P -> LDS (bf16) to re-layout from C/D layout into A-fragment layout
#pragma unroll
    for (int ct = 0; ct < 2; ++ct)
#pragma unroll
      for (int i = 0; i < 8; ++i)
        Ps[(i + 8 * half) * LDS_STRIDE + ct * 16 + l16] = f2bf(s[ct][i]);
    __syncthreads();

    FragU up;   // P as A fragment (16 rows x 32 keys)
#pragma unroll
    for (int j = 0; j < 8; ++j) {
      const int K = half * 16 + j * 2;
      up.d[j] = *(const unsigned int*)&Ps[l16 * LDS_STRIDE + K];
    }
    FragU uv[4];  // V^T B-fragments, gathered before the WMMAs
#pragma unroll
    for (int dt = 0; dt < 4; ++dt)
#pragma unroll
      for (int j = 0; j < 8; ++j)
        uv[dt].d[j] = *(const unsigned int*)&VT[(dt * 16 + l16) * LDS_STRIDE +
                                                half * 16 + 2 * j];

    // rescale O by exp(old_max - new_max), then O += P @ V
#pragma unroll
    for (int dt = 0; dt < 4; ++dt) {
#pragma unroll
      for (int i = 0; i < 8; ++i) o[dt][i] *= rsc[i];
      o[dt] = __builtin_amdgcn_wmma_f32_16x16x32_bf16(
          false, up.v, false, uv[dt].v, (short)0, o[dt], false, false);
    }
    __syncthreads();
  }

  // epilogue: y[b, t, h*64 + d] (bf16 for final GEMM); row t=0 forced to 0
#pragma unroll
  for (int dt = 0; dt < 4; ++dt)
#pragma unroll
    for (int i = 0; i < 8; ++i) {
      const int trow = t0 + i + 8 * half;
      const float den = rs[i];
      const float val = (trow == 0 || den <= 0.0f) ? 0.0f : o[dt][i] / den;
      ybf[((size_t)(b * T_ + trow)) * C_ + h * HS_ + dt * 16 + l16] = f2bf(val);
    }
}

// ---------------------------------------------------------------------------
// Launch
// ---------------------------------------------------------------------------
extern "C" void kernel_launch(void* const* d_in, const int* in_sizes, int n_in,
                              void* d_out, int out_size, void* d_ws, size_t ws_size,
                              hipStream_t stream) {
  (void)in_sizes; (void)n_in; (void)out_size; (void)ws_size;

  const float* x    = (const float*)d_in[0];
  const float* Wk   = (const float*)d_in[1];
  const float* Wv   = (const float*)d_in[2];
  const float* Wc   = (const float*)d_in[3];
  const float* beta = (const float*)d_in[4];
  const float* ksc  = (const float*)d_in[5];
  const float* vcf  = (const float*)d_in[6];
  const float* vsc  = (const float*)d_in[7];
  float* out = (float*)d_out;

  char* ws = (char*)d_ws;
  size_t off = 0;
  unsigned short* xbf  = (unsigned short*)(ws + off); off += (size_t)M_ * C_ * 2;
  unsigned short* wkbf = (unsigned short*)(ws + off); off += (size_t)C_ * C_ * 2;
  unsigned short* wvbf = (unsigned short*)(ws + off); off += (size_t)C_ * C_ * 2;
  unsigned short* wcbf = (unsigned short*)(ws + off); off += (size_t)C_ * C_ * 2;
  float*          Kraw = (float*)(ws + off);          off += (size_t)M_ * C_ * 4;
  float*          Vraw = (float*)(ws + off);          off += (size_t)M_ * C_ * 4;
  unsigned short* knb  = (unsigned short*)(ws + off); off += (size_t)M_ * C_ * 2;
  unsigned short* vnb  = (unsigned short*)(ws + off); off += (size_t)M_ * C_ * 2;
  unsigned short* ybf  = (unsigned short*)(ws + off); off += (size_t)M_ * C_ * 2;

  cm_f32_to_bf16<<<(M_ * C_ + 255) / 256, 256, 0, stream>>>(x,  xbf,  M_ * C_);
  cm_f32_to_bf16<<<(C_ * C_ + 255) / 256, 256, 0, stream>>>(Wk, wkbf, C_ * C_);
  cm_f32_to_bf16<<<(C_ * C_ + 255) / 256, 256, 0, stream>>>(Wv, wvbf, C_ * C_);
  cm_f32_to_bf16<<<(C_ * C_ + 255) / 256, 256, 0, stream>>>(Wc, wcbf, C_ * C_);

  dim3 gg(C_ / GT_MN, M_ / GT_MN);               // (8, 32)
  cm_gemm_bf16<1><<<gg, 256, 0, stream>>>(xbf, wkbf, Kraw, C_, C_);
  cm_gemm_bf16<1><<<gg, 256, 0, stream>>>(xbf, wvbf, Vraw, C_, C_);

  cm_decay<<<(B_ * NH_ * HS_) / 256, 256, 0, stream>>>(Kraw, beta);
  cm_knorm<<<(B_ * NH_ * T_) / 8, 256, 0, stream>>>(Kraw, ksc, knb);
  cm_vnorm<<<(B_ * NH_ * T_) / 8, 256, 0, stream>>>(Vraw, vcf, vsc, vnb);

  cm_attn<<<B_ * NH_ * (T_ / 16), 32, 0, stream>>>(knb, vnb, ybf);

  cm_gemm_bf16<0><<<gg, 256, 0, stream>>>(ybf, wcbf, out, C_, C_);
}
